// DecoderRNN_56822417326395
// MI455X (gfx1250) — compile-verified, matrix operand order checked by reading, not run on hardware
//
#include <hip/hip_runtime.h>
#include <hip/hip_bf16.h>
#include <math.h>

// Problem constants (match reference)
#define V_  32000
#define E_  512
#define H_  512
#define L_  2
#define B_  64
#define ML_ 64
#define T_  32

typedef __attribute__((ext_vector_type(16))) __bf16 v16bf;
typedef __attribute__((ext_vector_type(8)))  float  v8f;

// ---------------------------------------------------------------------------
// WMMA fragment helpers (wave32, V_WMMA_F32_16X16X32_BF16)
// A-matrix 16x32 bf16 layout (ISA 7.12.2): lanes 0-15 -> row M=lane, vec[0..7]
// = K 0..7, vec[8..15] = K 16..23 ; lanes 16-31 -> row M=lane-16, K 8..15 and
// 24..31.  Weights are stored row-major [N,K] (i.e. W^T), which makes the B
// fragment load identical with N in place of M.
// ---------------------------------------------------------------------------
__device__ inline v16bf load_tile16(const __bf16* __restrict__ base, int ld,
                                    int r0, int k0, int lane) {
    const __bf16* p = base + (size_t)(r0 + (lane & 15)) * ld + k0 + ((lane >> 4) << 3);
    v16bf v;
#pragma unroll
    for (int i = 0; i < 8; ++i) { v[i] = p[i]; v[i + 8] = p[16 + i]; }
    return v;
}

// One wave computes a full M=64 column strip (4 M-tiles x one 16-wide N tile),
// accumulating into acc[4].  B fragment loaded once per K-step, reused 4x.
__device__ inline void gemm_col64(const __bf16* __restrict__ A, int lda,
                                  const __bf16* __restrict__ W, int ldw,
                                  int n0, int K, int lane, v8f acc[4]) {
    for (int k = 0; k < K; k += 32) {
        v16bf b = load_tile16(W, ldw, n0, k, lane);
#pragma unroll
        for (int m = 0; m < 4; ++m) {
            v16bf a = load_tile16(A, lda, m * 16, k, lane);
            acc[m] = __builtin_amdgcn_wmma_f32_16x16x32_bf16(
                false, a, false, b, (short)0, acc[m], false, false);
        }
    }
}

__device__ inline float sigmoidf_(float x) { return 1.0f / (1.0f + expf(-x)); }

// ---------------------------------------------------------------------------
// Init kernels
// ---------------------------------------------------------------------------
__global__ void k_cvt_bf16(const float* __restrict__ src, __bf16* __restrict__ dst, int n) {
    for (int i = blockIdx.x * blockDim.x + threadIdx.x; i < n; i += gridDim.x * blockDim.x)
        dst[i] = (__bf16)src[i];
}

__global__ void k_init_state(const float* __restrict__ h0, const float* __restrict__ c0,
                             float* __restrict__ h, float* __restrict__ c,
                             __bf16* __restrict__ hbf, int n) {
    for (int i = blockIdx.x * blockDim.x + threadIdx.x; i < n; i += gridDim.x * blockDim.x) {
        float hv = h0[i];
        h[i] = hv; c[i] = c0[i]; hbf[i] = (__bf16)hv;
    }
}

// ---------------------------------------------------------------------------
// Step kernels
// ---------------------------------------------------------------------------

// Gather embedding row for this step's token and pack [emb | h_top] (bf16).
__global__ __launch_bounds__(256) void k_token_cat(
    int t, const int* __restrict__ cur, const int* __restrict__ gtruth,
    const float* __restrict__ emb, const __bf16* __restrict__ h_top_bf,
    __bf16* __restrict__ cat) {
    int b = blockIdx.x;
    int tok = (t == 0) ? cur[b] : gtruth[(t - 1) * B_ + b];
    const float* er = emb + (size_t)tok * E_;
    for (int e = threadIdx.x; e < E_; e += blockDim.x) {
        cat[b * (E_ + H_) + e]      = (__bf16)er[e];
        cat[b * (E_ + H_) + E_ + e] = h_top_bf[b * H_ + e];
    }
}

// Attention scores (64x64x1024 via WMMA) + row softmax. One block, 4 waves.
__global__ __launch_bounds__(128) void k_attn_softmax(
    const __bf16* __restrict__ cat, const __bf16* __restrict__ attnW,
    float* __restrict__ a_out) {
    __shared__ float sc[B_][ML_ + 1];
    int lane = threadIdx.x & 31, wave = threadIdx.x >> 5;
    int n0 = wave * 16;
    v8f acc[4] = {};
    gemm_col64(cat, E_ + H_, attnW, E_ + H_, n0, E_ + H_, lane, acc);
    int col = lane & 15, rb = (lane >> 4) * 8;
#pragma unroll
    for (int m = 0; m < 4; ++m)
#pragma unroll
        for (int r = 0; r < 8; ++r)
            sc[m * 16 + rb + r][n0 + col] = acc[m][r];
    __syncthreads();
    if (threadIdx.x < B_) {
        int b = threadIdx.x;
        float mx = sc[b][0];
        for (int l = 1; l < ML_; ++l) mx = fmaxf(mx, sc[b][l]);
        float s = 0.f;
        for (int l = 0; l < ML_; ++l) { float e = expf(sc[b][l] - mx); sc[b][l] = e; s += e; }
        float inv = 1.0f / s;
        for (int l = 0; l < ML_; ++l) a_out[b * ML_ + l] = sc[b][l] * inv;
    }
}

// ctx[b,h] = sum_l a[b,l] * enc[l,b,h]; written as bf16 into cat[:,512:].
__global__ __launch_bounds__(256) void k_ctx(
    const float* __restrict__ a, const float* __restrict__ enc,
    __bf16* __restrict__ cat) {
    int i = blockIdx.x * blockDim.x + threadIdx.x;    // 0 .. B*H-1
    int b = i >> 9, h = i & (H_ - 1);
    float s = 0.f;
    for (int l = 0; l < ML_; ++l)
        s += a[b * ML_ + l] * enc[((size_t)l * B_ + b) * H_ + h];
    cat[b * (E_ + H_) + E_ + h] = (__bf16)s;
}

// rnn_in = relu([emb|ctx] @ attn_out_W^T).  N=512 -> 32 tiles -> 8 blocks x 4 waves.
__global__ __launch_bounds__(128) void k_attn_out(
    const __bf16* __restrict__ cat, const __bf16* __restrict__ W,
    __bf16* __restrict__ x) {
    int lane = threadIdx.x & 31, wave = threadIdx.x >> 5;
    int n0 = (blockIdx.x * 4 + wave) * 16;
    v8f acc[4] = {};
    gemm_col64(cat, E_ + H_, W, E_ + H_, n0, E_ + H_, lane, acc);
    int col = lane & 15, rb = (lane >> 4) * 8;
#pragma unroll
    for (int m = 0; m < 4; ++m)
#pragma unroll
        for (int r = 0; r < 8; ++r) {
            float v = acc[m][r];
            x[(m * 16 + rb + r) * E_ + n0 + col] = (__bf16)(v > 0.f ? v : 0.f);
        }
}

// gates = x@Wi^T + h@Wh^T + bi + bh.  N=2048 -> 128 tiles -> 32 blocks x 4 waves.
__global__ __launch_bounds__(128) void k_lstm_gates(
    const __bf16* __restrict__ x, const __bf16* __restrict__ h,
    const __bf16* __restrict__ Wi, const __bf16* __restrict__ Wh,
    const float* __restrict__ bi, const float* __restrict__ bh,
    float* __restrict__ gates) {
    int lane = threadIdx.x & 31, wave = threadIdx.x >> 5;
    int n0 = (blockIdx.x * 4 + wave) * 16;
    v8f acc[4] = {};
    gemm_col64(x, E_, Wi, E_, n0, E_, lane, acc);
    gemm_col64(h, H_, Wh, H_, n0, H_, lane, acc);
    int col = lane & 15, rb = (lane >> 4) * 8;
    int j = n0 + col;
    float bias = bi[j] + bh[j];
#pragma unroll
    for (int m = 0; m < 4; ++m)
#pragma unroll
        for (int r = 0; r < 8; ++r)
            gates[(m * 16 + rb + r) * (4 * H_) + j] = acc[m][r] + bias;
}

// Element-wise LSTM cell update (PyTorch gate order i,f,g,o).
__global__ __launch_bounds__(256) void k_lstm_cell(
    const float* __restrict__ gates, float* __restrict__ c,
    float* __restrict__ h, __bf16* __restrict__ hbf) {
    int i = blockIdx.x * blockDim.x + threadIdx.x;    // 0 .. B*H-1
    int b = i >> 9, hh = i & (H_ - 1);
    const float* g = gates + (size_t)b * 4 * H_;
    float ig = sigmoidf_(g[hh]);
    float fg = sigmoidf_(g[H_ + hh]);
    float gg = tanhf(g[2 * H_ + hh]);
    float og = sigmoidf_(g[3 * H_ + hh]);
    float cn = fg * c[i] + ig * gg;
    float hn = og * tanhf(cn);
    c[i] = cn; h[i] = hn; hbf[i] = (__bf16)hn;
}

// scores = h_top @ gen_W^T + gen_b.  Register-blocked 4 M-tiles x 4 N-tiles per
// wave: A (h_top) traffic drops 4x vs 4x1 blocking; 16 WMMAs per 8 fragment
// loads.  N=32000 -> 500 n-groups of 64 -> 125 blocks x 4 waves.  Output is
// written once and never re-read -> nontemporal stores keep bf16 gen_W
// resident in the 192MB L2 across all 32 steps.
__global__ __launch_bounds__(128) void k_gen(
    const __bf16* __restrict__ h, const __bf16* __restrict__ Wg,
    const float* __restrict__ bg, float* __restrict__ out) {
    int lane = threadIdx.x & 31, wave = threadIdx.x >> 5;
    int n0 = (blockIdx.x * 4 + wave) * 64;            // 64-wide N group
    v8f acc[4][4] = {};
    for (int k = 0; k < H_; k += 32) {
        v16bf a[4], b[4];
#pragma unroll
        for (int m = 0; m < 4; ++m) a[m] = load_tile16(h, H_, m * 16, k, lane);
#pragma unroll
        for (int n = 0; n < 4; ++n) b[n] = load_tile16(Wg, H_, n0 + n * 16, k, lane);
#pragma unroll
        for (int n = 0; n < 4; ++n)
#pragma unroll
            for (int m = 0; m < 4; ++m)
                acc[n][m] = __builtin_amdgcn_wmma_f32_16x16x32_bf16(
                    false, a[m], false, b[n], (short)0, acc[n][m], false, false);
    }
    int col = lane & 15, rb = (lane >> 4) * 8;
#pragma unroll
    for (int n = 0; n < 4; ++n) {
        int v = n0 + n * 16 + col;
        float bias = bg[v];
#pragma unroll
        for (int m = 0; m < 4; ++m)
#pragma unroll
            for (int r = 0; r < 8; ++r)
                __builtin_nontemporal_store(acc[n][m][r] + bias,
                                            &out[(size_t)(m * 16 + rb + r) * V_ + v]);
    }
}

// Final h/c state -> tail of d_out.
__global__ void k_tail(const float* __restrict__ h, const float* __restrict__ c,
                       float* __restrict__ out, int n) {
    for (int i = blockIdx.x * blockDim.x + threadIdx.x; i < n; i += gridDim.x * blockDim.x)
        out[i] = (i < L_ * B_ * H_) ? h[i] : c[i - L_ * B_ * H_];
}

// ---------------------------------------------------------------------------
// Host launcher
// ---------------------------------------------------------------------------
extern "C" void kernel_launch(void* const* d_in, const int* in_sizes, int n_in,
                              void* d_out, int out_size, void* d_ws, size_t ws_size,
                              hipStream_t stream) {
    const int*   cur      = (const int*)  d_in[0];
    const float* h0       = (const float*)d_in[1];
    const float* c0       = (const float*)d_in[2];
    const float* enc      = (const float*)d_in[3];
    const int*   gtruth   = (const int*)  d_in[4];
    // d_in[5] = length (== T_)
    const float* emb      = (const float*)d_in[6];
    const float* attnW    = (const float*)d_in[7];
    const float* attnOutW = (const float*)d_in[8];
    const float* lstmWi   = (const float*)d_in[9];
    const float* lstmWh   = (const float*)d_in[10];
    const float* lstmBi   = (const float*)d_in[11];
    const float* lstmBh   = (const float*)d_in[12];
    const float* genW     = (const float*)d_in[13];
    const float* genB     = (const float*)d_in[14];
    (void)in_sizes; (void)n_in; (void)out_size; (void)ws_size;

    // Workspace layout (bytes) -- ~41.7 MB, bf16 weights stay L2-resident.
    char* ws = (char*)d_ws;
    size_t off = 0;
    __bf16* genW_bf  = (__bf16*)(ws + off); off += (size_t)V_ * H_ * 2;
    __bf16* Wi_bf    = (__bf16*)(ws + off); off += (size_t)L_ * 4 * H_ * E_ * 2;
    __bf16* Wh_bf    = (__bf16*)(ws + off); off += (size_t)L_ * 4 * H_ * H_ * 2;
    __bf16* aoW_bf   = (__bf16*)(ws + off); off += (size_t)E_ * (E_ + H_) * 2;
    __bf16* attnW_bf = (__bf16*)(ws + off); off += (size_t)ML_ * (E_ + H_) * 2;
    float*  h_f      = (float*)(ws + off);  off += (size_t)L_ * B_ * H_ * 4;
    float*  c_f      = (float*)(ws + off);  off += (size_t)L_ * B_ * H_ * 4;
    __bf16* h_bf     = (__bf16*)(ws + off); off += (size_t)L_ * B_ * H_ * 2;
    __bf16* cat_bf   = (__bf16*)(ws + off); off += (size_t)B_ * (E_ + H_) * 2;
    __bf16* x_bf     = (__bf16*)(ws + off); off += (size_t)B_ * E_ * 2;
    float*  a_f      = (float*)(ws + off);  off += (size_t)B_ * ML_ * 4;
    float*  gates_f  = (float*)(ws + off);  off += (size_t)B_ * 4 * H_ * 4;

    float* out = (float*)d_out;

    // One-time (per call) weight conversion fp32 -> bf16.
    auto cvt = [&](const float* s, __bf16* d, int n) {
        int grid = (n + 255) / 256; if (grid > 8192) grid = 8192;
        k_cvt_bf16<<<grid, 256, 0, stream>>>(s, d, n);
    };
    cvt(genW,     genW_bf,  V_ * H_);
    cvt(lstmWi,   Wi_bf,    L_ * 4 * H_ * E_);
    cvt(lstmWh,   Wh_bf,    L_ * 4 * H_ * H_);
    cvt(attnOutW, aoW_bf,   E_ * (E_ + H_));
    cvt(attnW,    attnW_bf, ML_ * (E_ + H_));
    k_init_state<<<(L_ * B_ * H_ + 255) / 256, 256, 0, stream>>>(
        h0, c0, h_f, c_f, h_bf, L_ * B_ * H_);

    __bf16* h_top_bf = h_bf + (size_t)(L_ - 1) * B_ * H_;

    for (int t = 0; t < T_; ++t) {
        k_token_cat<<<B_, 256, 0, stream>>>(t, cur, gtruth, emb, h_top_bf, cat_bf);
        k_attn_softmax<<<1, 128, 0, stream>>>(cat_bf, attnW_bf, a_f);
        k_ctx<<<(B_ * H_) / 256, 256, 0, stream>>>(a_f, enc, cat_bf);
        k_attn_out<<<(E_ / 16) / 4, 128, 0, stream>>>(cat_bf, aoW_bf, x_bf);
        for (int l = 0; l < L_; ++l) {
            const __bf16* xin = (l == 0) ? x_bf : h_bf;   // layer1 input = new h of layer0
            __bf16* hl_bf = h_bf + (size_t)l * B_ * H_;
            float*  hl_f  = h_f  + (size_t)l * B_ * H_;
            float*  cl_f  = c_f  + (size_t)l * B_ * H_;
            k_lstm_gates<<<(4 * H_ / 16) / 4, 128, 0, stream>>>(
                xin, hl_bf,
                Wi_bf + (size_t)l * 4 * H_ * E_, Wh_bf + (size_t)l * 4 * H_ * H_,
                lstmBi + (size_t)l * 4 * H_, lstmBh + (size_t)l * 4 * H_, gates_f);
            k_lstm_cell<<<(B_ * H_) / 256, 256, 0, stream>>>(gates_f, cl_f, hl_f, hl_bf);
        }
        k_gen<<<(V_ / 64) / 4, 128, 0, stream>>>(
            h_top_bf, genW_bf, genB, out + (size_t)t * B_ * V_);
    }

    k_tail<<<(2 * L_ * B_ * H_ + 255) / 256, 256, 0, stream>>>(
        h_f, c_f, out + (size_t)T_ * B_ * V_, 2 * L_ * B_ * H_);
}
